// LocationExpertRouter_27934467293828
// MI455X (gfx1250) — matrix operand order repaired and checked by reading, not compiled, not run on hardware
//
#include <hip/hip_runtime.h>
#include <hip/hip_bf16.h>

#define NUM_EXPERTS 8
#define D_MODEL     1024
#define VOCAB       32000
#define BATCH       4096

#define M_TILE  64
#define N_TILE  128
#define K_STAGE 64            // staged per barrier round (2 WMMA k-steps)
#define K_WMMA  32
#define LDS_K   (K_STAGE + 8) // row padding: 144B stride, 16B aligned

typedef __attribute__((ext_vector_type(16))) __bf16 v16bf;
typedef __attribute__((ext_vector_type(8)))  __bf16 v8bf;
typedef __attribute__((ext_vector_type(8)))  float  v8f;

// pack two f32 -> two bf16 (RNE) in one VALU op (pure: let compiler schedule/CSE)
__device__ __forceinline__ unsigned cvt_pk_bf16(float a, float b) {
    unsigned d;
    asm("v_cvt_pk_bf16_f32 %0, %1, %2" : "=v"(d) : "v"(a), "v"(b));
    return d;
}

// ---------------- routing ----------------

__global__ void ler_zero_counts(int* counts) {
    if (threadIdx.x < NUM_EXPERTS) counts[threadIdx.x] = 0;
}

__global__ void ler_route(const long long* __restrict__ ptrs,
                          int* __restrict__ counts,
                          int* __restrict__ rowlist) {
    int r = blockIdx.x * blockDim.x + threadIdx.x;
    if (r < BATCH) {
        int e = (int)(ptrs[r] & (long long)(NUM_EXPERTS - 1)); // ptrs >= 0
        int slot = atomicAdd(&counts[e], 1);
        rowlist[e * BATCH + slot] = r;
    }
}

// ---------------- grouped GEMM ----------------
// grid: x = VOCAB/N_TILE (250), y = BATCH/M_TILE (64), z = experts (8)
// block: 256 threads = 8 wave32s arranged 2 (M) x 4 (N); wave tile 32x32.

__global__ __launch_bounds__(256)
void ler_gemm(const float* __restrict__ x,
              const float* __restrict__ W,
              const float* __restrict__ b,
              const int* __restrict__ counts,
              const int* __restrict__ rowlist,
              float* __restrict__ out) {
    const int e   = blockIdx.z;
    const int cnt = counts[e];
    const int rowStart = blockIdx.y * M_TILE;
    if (rowStart >= cnt) return;                // block-uniform early exit
    const int n0 = blockIdx.x * N_TILE;

    const int tid  = threadIdx.x;
    const int lane = tid & 31;
    const int wid  = tid >> 5;
    const int wm   = wid & 1;                   // 0..1 along M
    const int wn   = wid >> 1;                  // 0..3 along N

    __shared__ __attribute__((aligned(64))) unsigned short As[M_TILE][LDS_K]; // [m][k] bf16 bits
    __shared__ __attribute__((aligned(64))) unsigned short Bs[N_TILE][LDS_K]; // [n][k] bf16 bits
    __shared__ int rowsS[M_TILE];

    if (tid < M_TILE) {
        int g = rowStart + tid;
        rowsS[tid] = (g < cnt) ? rowlist[e * BATCH + g] : -1;
    }
    __syncthreads();

    const float* __restrict__ wBase = W + (size_t)e * VOCAB * D_MODEL;

    // ---- hoisted, k-invariant staging pointers ----
    // Each thread owns float4 slots at column kq of 4 A-rows and 8 B-rows.
    const int kq   = (tid & 15) * 4;            // 0..60
    const int rBase = tid >> 4;                 // 0..15

    const float*    aP[4]; unsigned short* aL[4];
    #pragma unroll
    for (int i = 0; i < 4; ++i) {
        int m   = rBase + i * 16;
        int row = rowsS[m];
        int rs  = row < 0 ? 0 : row;            // clamp: padded rows read row 0;
        aP[i] = x + (size_t)rs * D_MODEL + kq;  // their outputs are never stored
        aL[i] = &As[m][kq];
    }
    const float*    bP[8]; unsigned short* bL[8];
    #pragma unroll
    for (int i = 0; i < 8; ++i) {
        int n = rBase + i * 16;
        bP[i] = wBase + (size_t)(n0 + n) * D_MODEL + kq;
        bL[i] = &Bs[n][kq];
    }

    v8f acc[2][2] = {};                         // 32x32 per wave, f32 accumulate

    // ---- software pipeline: regs hold stage k0 while LDS serves k0-64 ----
    float4 aReg[4], bReg[8];
    #pragma unroll
    for (int i = 0; i < 4; ++i) aReg[i] = *(const float4*)aP[i];
    #pragma unroll
    for (int i = 0; i < 8; ++i) bReg[i] = *(const float4*)bP[i];

    for (int k0 = 0; k0 < D_MODEL; k0 += K_STAGE) {
        // commit staged registers to LDS as packed bf16 (ds_store_b64)
        #pragma unroll
        for (int i = 0; i < 4; ++i) {
            uint2 pk = { cvt_pk_bf16(aReg[i].x, aReg[i].y),
                         cvt_pk_bf16(aReg[i].z, aReg[i].w) };
            *(uint2*)aL[i] = pk;
        }
        #pragma unroll
        for (int i = 0; i < 8; ++i) {
            uint2 pk = { cvt_pk_bf16(bReg[i].x, bReg[i].y),
                         cvt_pk_bf16(bReg[i].z, bReg[i].w) };
            *(uint2*)bL[i] = pk;
        }
        __syncthreads();

        // issue next stage's global loads before the WMMA burst (latency overlap)
        if (k0 + K_STAGE < D_MODEL) {
            #pragma unroll
            for (int i = 0; i < 4; ++i) aReg[i] = *(const float4*)(aP[i] + k0 + K_STAGE);
            #pragma unroll
            for (int i = 0; i < 8; ++i) bReg[i] = *(const float4*)(bP[i] + k0 + K_STAGE);
            __builtin_prefetch(bP[0] + k0 + 2 * K_STAGE, 0, 0);  // warm L2 ahead
        }

        // ---- 2 WMMA k-steps per staged chunk ----
        #pragma unroll
        for (int kk = 0; kk < K_STAGE; kk += K_WMMA) {
            // A 16x32 bf16 (ISA 7.12.2): m = lane&15; lanes<16 hold K {0..7,16..23},
            // lanes>=16 hold {8..15,24..31}  -> two 8-elem chunks per lane
            const int aBase = kk + ((lane >> 4) << 3);
            // B 32x16 bf16: n = lane&15; lanes<16 hold K {0..7,16..23}, lanes>=16 hold {8..15,24..31}
            const int bBase = kk + ((lane >> 4) << 3);

            v16bf afrag[2], bfrag[2];
            #pragma unroll
            for (int im = 0; im < 2; ++im) {
                int m = wm * 32 + im * 16 + (lane & 15);
                v8bf lo = *(const v8bf*)&As[m][aBase];
                v8bf hi = *(const v8bf*)&As[m][aBase + 16];
                #pragma unroll
                for (int j = 0; j < 8; ++j) { afrag[im][j] = lo[j]; afrag[im][j + 8] = hi[j]; }
            }
            #pragma unroll
            for (int in = 0; in < 2; ++in) {
                int n = wn * 32 + in * 16 + (lane & 15);
                v8bf lo = *(const v8bf*)&Bs[n][bBase];
                v8bf hi = *(const v8bf*)&Bs[n][bBase + 16];
                #pragma unroll
                for (int j = 0; j < 8; ++j) { bfrag[in][j] = lo[j]; bfrag[in][j + 8] = hi[j]; }
            }

            #pragma unroll
            for (int im = 0; im < 2; ++im)
                #pragma unroll
                for (int in = 0; in < 2; ++in)
                    acc[im][in] = __builtin_amdgcn_wmma_f32_16x16x32_bf16(
                        false, afrag[im], false, bfrag[in],
                        (short)0, acc[im][in], false, false);
        }
        __syncthreads();
    }

    // ---- epilogue: bias + predicated scatter by gathered row ----
    const float* __restrict__ bias = b + (size_t)e * VOCAB;
    #pragma unroll
    for (int im = 0; im < 2; ++im) {
        int mBase = wm * 32 + im * 16;
        #pragma unroll
        for (int in = 0; in < 2; ++in) {
            int nGlob = n0 + wn * 32 + in * 16 + (lane & 15);
            float bv = bias[nGlob];
            v8f a = acc[im][in];
            #pragma unroll
            for (int r = 0; r < 8; ++r) {
                int m   = mBase + r + ((lane >> 4) << 3); // C/D: lanes>=16 carry M+8
                int row = rowsS[m];
                if (row >= 0)
                    out[(size_t)row * VOCAB + nGlob] = a[r] + bv;
            }
        }
    }
}

// ---------------- launcher ----------------

extern "C" void kernel_launch(void* const* d_in, const int* in_sizes, int n_in,
                              void* d_out, int out_size, void* d_ws, size_t ws_size,
                              hipStream_t stream) {
    const float*     x    = (const float*)d_in[0];
    const long long* ptrs = (const long long*)d_in[1];
    const float*     W    = (const float*)d_in[2];
    const float*     b    = (const float*)d_in[3];
    float*           out  = (float*)d_out;

    int* counts  = (int*)d_ws;                       // 8 ints
    int* rowlist = (int*)d_ws + 16;                  // 8 * 4096 ints

    ler_zero_counts<<<1, 32, 0, stream>>>(counts);
    ler_route<<<(BATCH + 255) / 256, 256, 0, stream>>>(ptrs, counts, rowlist);

    dim3 grid(VOCAB / N_TILE, BATCH / M_TILE, NUM_EXPERTS);
    ler_gemm<<<grid, 256, 0, stream>>>(x, W, b, counts, rowlist, out);
}
